// NonLocalBlock_73521250173422
// MI455X (gfx1250) — matrix operand order repaired and checked by reading, not compiled, not run on hardware
//
#include <hip/hip_runtime.h>
#include <hip/hip_bf16.h>

// Problem constants (from the reference)
#define KDIM   4096
#define CDIM   768
#define CHDIM  384
#define HDIM   256
#define EPSV   1e-5f

typedef __attribute__((ext_vector_type(16))) __bf16 v16bf;
typedef __attribute__((ext_vector_type(8)))  __bf16 v8bf;
typedef __attribute__((ext_vector_type(8)))  float  v8f;

// ---------------------------------------------------------------------------
// WMMA helpers: D = A(16x32 bf16) * B(32x16 bf16) + C(16x16 f32)
// A fragment (ISA 7.12.2, 16-bit A 16x32): lane L holds row M=L%16;
//   lanes 0-15: K=0..7 (v0-3) and K=16..23 (v4-7); lanes 16-31: K=8..15, 24..31
// B fragment (32x16): N = lane%16; lanes 0-15 hold K=0..7, 16..23; lanes 16-31 K=8..15, 24..31
// C/D fragment: VGPR r, lane L -> (M = r + (L>=16)*8, N = L%16)
// ---------------------------------------------------------------------------
static __device__ __forceinline__ v8f wmma_bf16(v16bf a, v16bf b, v8f c) {
  return __builtin_amdgcn_wmma_f32_16x16x32_bf16(false, a, false, b, (short)0, c,
                                                 false, false);
}

static __device__ __forceinline__ v16bf load_a_frag(const __bf16* __restrict__ A,
                                                    int lda, int m0, int k0, int lane) {
  const __bf16* p = A + (size_t)(m0 + (lane & 15)) * lda + k0 + ((lane >> 4) << 3);
  v8bf lo = *(const v8bf*)(p);        // 16B load
  v8bf hi = *(const v8bf*)(p + 16);   // 16B load
  return __builtin_shufflevector(lo, hi, 0,1,2,3,4,5,6,7,8,9,10,11,12,13,14,15);
}

static __device__ __forceinline__ v16bf load_b_frag(const __bf16* __restrict__ BT,
                                                    int ldb, int n0, int k0, int lane) {
  const __bf16* p = BT + (size_t)(n0 + (lane & 15)) * ldb + k0 + ((lane >> 4) << 3);
  v8bf lo = *(const v8bf*)(p);        // 16B load
  v8bf hi = *(const v8bf*)(p + 16);   // 16B load
  return __builtin_shufflevector(lo, hi, 0,1,2,3,4,5,6,7,8,9,10,11,12,13,14,15);
}

// 32x32 output tile per wave: c[0]=(m0,n0) c[1]=(m0,n0+16) c[2]=(m0+16,n0) c[3]=+16,+16
static __device__ __forceinline__ void wmma_tile32(const __bf16* __restrict__ A, int lda,
                                                   const __bf16* __restrict__ BT, int ldb,
                                                   int kdim, int m0, int n0, int lane,
                                                   v8f c[4]) {
  v8f z = {};
  c[0] = z; c[1] = z; c[2] = z; c[3] = z;
  for (int k0 = 0; k0 < kdim; k0 += 32) {
    v16bf a0 = load_a_frag(A, lda, m0,      k0, lane);
    v16bf a1 = load_a_frag(A, lda, m0 + 16, k0, lane);
    v16bf b0 = load_b_frag(BT, ldb, n0,      k0, lane);
    v16bf b1 = load_b_frag(BT, ldb, n0 + 16, k0, lane);
    c[0] = wmma_bf16(a0, b0, c[0]);
    c[1] = wmma_bf16(a0, b1, c[1]);
    c[2] = wmma_bf16(a1, b0, c[2]);
    c[3] = wmma_bf16(a1, b1, c[3]);
  }
}

#define WAVE_TILE_SETUP(Mv, Nv)                                          \
  int lane = threadIdx.x & 31;                                           \
  int wave = blockIdx.x * (blockDim.x >> 5) + (threadIdx.x >> 5);        \
  const int tn = (Nv) >> 5;                                              \
  if (wave >= ((Mv) >> 5) * tn) return; /* wave-uniform: EXEC stays ~0 */\
  int m0 = (wave / tn) << 5;                                             \
  int n0 = (wave % tn) << 5;

// ---------------------------------------------------------------------------
// Small prep kernels
// ---------------------------------------------------------------------------
__global__ void cvt_bf16(const float* __restrict__ in, __bf16* __restrict__ out, int n) {
  int i = blockIdx.x * blockDim.x + threadIdx.x;
  if (i < n) out[i] = (__bf16)in[i];
}

// xb[j*768 + c] = bf16(x[c*4096 + j])  (x^T for B-fragments of the projections)
__global__ void pack_xT(const float* __restrict__ x, __bf16* __restrict__ xb) {
  int o = blockIdx.x * blockDim.x + threadIdx.x;   // 768*4096 threads
  int j = o / CDIM, c = o % CDIM;
  xb[o] = (__bf16)x[(size_t)c * KDIM + j];
}

// ha[j] = (h0 . Wh0[j,:] + bh0[j]) * (h1 . Wh1[j,:] + bh1[j])
__global__ void ha_kernel(const float* __restrict__ h0, const float* __restrict__ h1,
                          const float* __restrict__ Wh0, const float* __restrict__ bh0,
                          const float* __restrict__ Wh1, const float* __restrict__ bh1,
                          float* __restrict__ ha) {
  int j = blockIdx.x * blockDim.x + threadIdx.x;
  if (j >= KDIM) return;
  float s0 = bh0[j], s1 = bh1[j];
  for (int c = 0; c < HDIM; ++c) {
    s0 = fmaf(h0[c], Wh0[(size_t)j * HDIM + c], s0);
    s1 = fmaf(h1[c], Wh1[(size_t)j * HDIM + c], s1);
  }
  ha[j] = s0 * s1;
}

// ---------------------------------------------------------------------------
// GEMM 1: y = BN(W @ x + b); W(384x768) bf16, xT(4096x768) bf16.
// mode 0/1 (q,k): store y^T -> outb[n*384 + m]       (4096x384 bf16)
// mode 2   (v)  : store reshape-transpose v_r^T      (384x4096 bf16)
// ---------------------------------------------------------------------------
__global__ __launch_bounds__(256)
void gemm_proj(const __bf16* __restrict__ Wb, const __bf16* __restrict__ xb,
               const float* __restrict__ bias, const float* __restrict__ g,
               const float* __restrict__ beta, const float* __restrict__ mean,
               const float* __restrict__ var, __bf16* __restrict__ outb, int mode) {
  WAVE_TILE_SETUP(CHDIM, KDIM)
  v8f c[4];
  wmma_tile32(Wb, CDIM, xb, CDIM, CDIM, m0, n0, lane, c);
#pragma unroll
  for (int ai = 0; ai < 2; ++ai)
#pragma unroll
    for (int bi = 0; bi < 2; ++bi) {
      int ncol  = n0 + bi * 16 + (lane & 15);
      int rbase = m0 + ai * 16 + ((lane >> 4) << 3);
      v8f f = c[ai * 2 + bi];
#pragma unroll
      for (int r = 0; r < 8; ++r) {
        int row = rbase + r;
        float sc = g[row] * rsqrtf(var[row] + EPSV);
        float y  = (f[r] + bias[row] - mean[row]) * sc + beta[row];
        if (mode == 2) {
          size_t fl = (size_t)row * KDIM + ncol;              // flat index in v
          outb[(fl % CHDIM) * (size_t)KDIM + (fl / CHDIM)] = (__bf16)y;
        } else {
          outb[(size_t)ncol * CHDIM + row] = (__bf16)y;
        }
      }
    }
}

// ---------------------------------------------------------------------------
// GEMM 2: S[i,j] = (q^T k)[i,j] * ha[j]   -> f32 into d_out attn region
// A = qT (4096x384), BT = kT (4096x384), Kd = 384
// ---------------------------------------------------------------------------
__global__ __launch_bounds__(256)
void gemm_qk(const __bf16* __restrict__ qb, const __bf16* __restrict__ kb,
             const float* __restrict__ ha, float* __restrict__ S) {
  WAVE_TILE_SETUP(KDIM, KDIM)
  v8f c[4];
  wmma_tile32(qb, CHDIM, kb, CHDIM, CHDIM, m0, n0, lane, c);
#pragma unroll
  for (int ai = 0; ai < 2; ++ai)
#pragma unroll
    for (int bi = 0; bi < 2; ++bi) {
      int ncol  = n0 + bi * 16 + (lane & 15);
      int rbase = m0 + ai * 16 + ((lane >> 4) << 3);
      float hav = ha[ncol];
      v8f f = c[ai * 2 + bi];
#pragma unroll
      for (int r = 0; r < 8; ++r)
        S[(size_t)(rbase + r) * KDIM + ncol] = f[r] * hav;
    }
}

// ---------------------------------------------------------------------------
// Row softmax over S (in d_out, in place) + bf16 copy for the AV GEMM
// ---------------------------------------------------------------------------
__global__ __launch_bounds__(256)
void softmax_rows(float* __restrict__ S, __bf16* __restrict__ attn_b) {
  __shared__ float red[256];
  int row = blockIdx.x, tid = threadIdx.x;
  float* p = S + (size_t)row * KDIM;
  float vals[16], mx = -3.4e38f;
#pragma unroll
  for (int i = 0; i < 16; ++i) { vals[i] = p[tid + i * 256]; mx = fmaxf(mx, vals[i]); }
  red[tid] = mx; __syncthreads();
  for (int s = 128; s > 0; s >>= 1) {
    if (tid < s) red[tid] = fmaxf(red[tid], red[tid + s]);
    __syncthreads();
  }
  mx = red[0]; __syncthreads();
  float sum = 0.f;
#pragma unroll
  for (int i = 0; i < 16; ++i) { vals[i] = __expf(vals[i] - mx); sum += vals[i]; }
  red[tid] = sum; __syncthreads();
  for (int s = 128; s > 0; s >>= 1) {
    if (tid < s) red[tid] += red[tid + s];
    __syncthreads();
  }
  float inv = 1.f / red[0];
#pragma unroll
  for (int i = 0; i < 16; ++i) {
    float a = vals[i] * inv;
    p[tid + i * 256] = a;
    attn_b[(size_t)row * KDIM + tid + i * 256] = (__bf16)a;
  }
}

// ---------------------------------------------------------------------------
// GEMM 3: out_av = attn @ v_r  (4096x384), Kd = 4096
// Epilogue scatters the reshape-transpose g^T: gb[j*384 + c] (4096x384 bf16)
// ---------------------------------------------------------------------------
__global__ __launch_bounds__(256)
void gemm_av(const __bf16* __restrict__ attn_b, const __bf16* __restrict__ vb,
             __bf16* __restrict__ gb) {
  WAVE_TILE_SETUP(KDIM, CHDIM)
  v8f c[4];
  wmma_tile32(attn_b, KDIM, vb, KDIM, KDIM, m0, n0, lane, c);
#pragma unroll
  for (int ai = 0; ai < 2; ++ai)
#pragma unroll
    for (int bi = 0; bi < 2; ++bi) {
      int ncol  = n0 + bi * 16 + (lane & 15);
      int rbase = m0 + ai * 16 + ((lane >> 4) << 3);
      v8f f = c[ai * 2 + bi];
#pragma unroll
      for (int r = 0; r < 8; ++r) {
        size_t fl = (size_t)(rbase + r) * CHDIM + ncol;       // flat of (K,CH)
        gb[(fl & (KDIM - 1)) * (size_t)CHDIM + (fl >> 12)] = (__bf16)f[r];
      }
    }
}

// ---------------------------------------------------------------------------
// GEMM 4: res = x + Wf @ g + bf   (768x4096), Kd = 384 -> f32 d_out
// ---------------------------------------------------------------------------
__global__ __launch_bounds__(256)
void gemm_final(const __bf16* __restrict__ Wfb, const __bf16* __restrict__ gb,
                const float* __restrict__ x, const float* __restrict__ bfv,
                float* __restrict__ res) {
  WAVE_TILE_SETUP(CDIM, KDIM)
  v8f c[4];
  wmma_tile32(Wfb, CHDIM, gb, CHDIM, CHDIM, m0, n0, lane, c);
#pragma unroll
  for (int ai = 0; ai < 2; ++ai)
#pragma unroll
    for (int bi = 0; bi < 2; ++bi) {
      int ncol  = n0 + bi * 16 + (lane & 15);
      int rbase = m0 + ai * 16 + ((lane >> 4) << 3);
      v8f f = c[ai * 2 + bi];
#pragma unroll
      for (int r = 0; r < 8; ++r) {
        size_t idx = (size_t)(rbase + r) * KDIM + ncol;
        res[idx] = x[idx] + f[r] + bfv[rbase + r];
      }
    }
}

// ---------------------------------------------------------------------------
extern "C" void kernel_launch(void* const* d_in, const int* in_sizes, int n_in,
                              void* d_out, int out_size, void* d_ws, size_t ws_size,
                              hipStream_t stream) {
  const float* x    = (const float*)d_in[0];
  const float* h0   = (const float*)d_in[1];
  const float* h1   = (const float*)d_in[2];
  const float* Wq   = (const float*)d_in[3];  const float* bq  = (const float*)d_in[4];
  const float* Wk   = (const float*)d_in[5];  const float* bk  = (const float*)d_in[6];
  const float* Wv   = (const float*)d_in[7];  const float* bv  = (const float*)d_in[8];
  const float* Wf   = (const float*)d_in[9];  const float* bfv = (const float*)d_in[10];
  const float* gq = (const float*)d_in[11], *betq = (const float*)d_in[12],
             * mq = (const float*)d_in[13], *vq   = (const float*)d_in[14];
  const float* gk = (const float*)d_in[15], *betk = (const float*)d_in[16],
             * mk = (const float*)d_in[17], *vk   = (const float*)d_in[18];
  const float* gv = (const float*)d_in[19], *betv = (const float*)d_in[20],
             * mv = (const float*)d_in[21], *vv   = (const float*)d_in[22];
  const float* Wh0 = (const float*)d_in[23], *bh0 = (const float*)d_in[24];
  const float* Wh1 = (const float*)d_in[25], *bh1 = (const float*)d_in[26];

  float* out_res  = (float*)d_out;                         // 768*4096
  float* out_attn = out_res + (size_t)CDIM * KDIM;         // 4096*4096 (also scratch)

  // Workspace carve-up (256B aligned): ~54 MB total
  char*  ws  = (char*)d_ws;
  size_t off = 0;
  auto carve = [&](size_t bytes) {
    void* p = ws + off;
    off = (off + bytes + 255) & ~(size_t)255;
    return p;
  };
  __bf16* xb   = (__bf16*)carve((size_t)CDIM * KDIM * 2);   // x^T      (4096x768)
  __bf16* Wqb  = (__bf16*)carve((size_t)CHDIM * CDIM * 2);
  __bf16* Wkb  = (__bf16*)carve((size_t)CHDIM * CDIM * 2);
  __bf16* Wvb  = (__bf16*)carve((size_t)CHDIM * CDIM * 2);
  __bf16* Wfb  = (__bf16*)carve((size_t)CDIM * CHDIM * 2);
  __bf16* qb   = (__bf16*)carve((size_t)KDIM * CHDIM * 2);  // q^T      (4096x384)
  __bf16* kb   = (__bf16*)carve((size_t)KDIM * CHDIM * 2);  // k^T
  __bf16* vb   = (__bf16*)carve((size_t)CHDIM * KDIM * 2);  // v_r^T    (384x4096)
  __bf16* gb   = (__bf16*)carve((size_t)KDIM * CHDIM * 2);  // g^T      (4096x384)
  __bf16* atb  = (__bf16*)carve((size_t)KDIM * KDIM * 2);   // attn bf16
  float*  ha   = (float*)carve((size_t)KDIM * 4);
  (void)ws_size; (void)in_sizes; (void)n_in; (void)out_size;

  const int wN = CHDIM * CDIM;                              // 294912
  cvt_bf16<<<(wN + 255) / 256, 256, 0, stream>>>(Wq, Wqb, wN);
  cvt_bf16<<<(wN + 255) / 256, 256, 0, stream>>>(Wk, Wkb, wN);
  cvt_bf16<<<(wN + 255) / 256, 256, 0, stream>>>(Wv, Wvb, wN);
  cvt_bf16<<<(wN + 255) / 256, 256, 0, stream>>>(Wf, Wfb, wN);
  pack_xT<<<(CDIM * KDIM) / 256, 256, 0, stream>>>(x, xb);
  ha_kernel<<<KDIM / 256, 256, 0, stream>>>(h0, h1, Wh0, bh0, Wh1, bh1, ha);

  // q, k, v projections + BN (tiles: 12*128 = 1536 waves -> 192 blocks)
  gemm_proj<<<192, 256, 0, stream>>>(Wqb, xb, bq, gq, betq, mq, vq, qb, 0);
  gemm_proj<<<192, 256, 0, stream>>>(Wkb, xb, bk, gk, betk, mk, vk, kb, 1);
  gemm_proj<<<192, 256, 0, stream>>>(Wvb, xb, bv, gv, betv, mv, vv, vb, 2);

  // qk logits * ha -> attn region of d_out (128*128 = 16384 waves -> 2048 blocks)
  gemm_qk<<<2048, 256, 0, stream>>>(qb, kb, ha, out_attn);

  // softmax in place (f32 output) + bf16 copy
  softmax_rows<<<KDIM, 256, 0, stream>>>(out_attn, atb);

  // attn @ v_r  (128*12 = 1536 waves -> 192 blocks)
  gemm_av<<<192, 256, 0, stream>>>(atb, vb, gb);

  // Wf @ g + bf + x  (24*128 = 3072 waves -> 384 blocks)
  gemm_final<<<384, 256, 0, stream>>>(Wfb, gb, x, bfv, out_res);
}
